// BlockSparseAttention_13932873908391
// MI455X (gfx1250) — matrix operand order repaired and checked by reading, not compiled, not run on hardware
//
#include <hip/hip_runtime.h>
#include <hip/hip_bf16.h>

// ---------------------------------------------------------------------------
// Problem constants (from reference)
// ---------------------------------------------------------------------------
#define BB   2
#define SS   4096
#define HID  2048
#define HH   16
#define DD   128
#define BSZ  64
#define NBL  64
#define MROWS (BB * SS)              // 8192
#define SCALE 0.08838834764831845f  // 128^-0.5

// ---------------------------------------------------------------------------
// Types for WMMA
// ---------------------------------------------------------------------------
typedef __attribute__((ext_vector_type(16))) __bf16         v16bf;
typedef __attribute__((ext_vector_type(8)))  float          v8f;
typedef __attribute__((ext_vector_type(8)))  unsigned short su8;
typedef __attribute__((ext_vector_type(16))) unsigned short su16;

// round-to-nearest-even f32 -> bf16 (bit pattern as ushort)
__device__ __forceinline__ unsigned short f2bf(float f) {
    unsigned int u = __builtin_bit_cast(unsigned int, f);
    u += 0x7FFFu + ((u >> 16) & 1u);
    return (unsigned short)(u >> 16);
}

// Assemble a 16-half WMMA fragment from two aligned 16-byte chunks.
__device__ __forceinline__ v16bf ldfrag(const unsigned short* p0,
                                        const unsigned short* p1) {
    su8 lo = *(const su8*)p0;
    su8 hi = *(const su8*)p1;
    su16 w;
#pragma unroll
    for (int i = 0; i < 8; ++i) { w[i] = lo[i]; w[i + 8] = hi[i]; }
    return __builtin_bit_cast(v16bf, w);
}

__device__ __forceinline__ v8f wmma_bf16(v16bf a, v16bf b, v8f c) {
    return __builtin_amdgcn_wmma_f32_16x16x32_bf16(
        /*neg_a=*/false, a, /*neg_b=*/false, b,
        /*c_mod=*/(short)0, c, /*reuse_a=*/false, /*reuse_b=*/false);
}

// ---------------------------------------------------------------------------
// Routing kernels (tiny): column mean over S, then 16-wide router
// ---------------------------------------------------------------------------
#define NCHUNK 16
__global__ __launch_bounds__(256)
void colmean_part_kernel(const float* __restrict__ hs, float* __restrict__ part) {
    int t = blockIdx.x * 256 + threadIdx.x;      // 0 .. B*HID-1
    int z = blockIdx.y;                           // S chunk
    int b = t >> 11;                              // HID = 2048
    int c = t & (HID - 1);
    const float* p = hs + (size_t)b * SS * HID + (size_t)(z * (SS / NCHUNK)) * HID + c;
    float s = 0.f;
    for (int i = 0; i < SS / NCHUNK; ++i) s += p[(size_t)i * HID];
    part[(size_t)z * (BB * HID) + t] = s;
}

__global__ __launch_bounds__(256)
void colmean_reduce_kernel(const float* __restrict__ part, float* __restrict__ meanf) {
    int t = blockIdx.x * 256 + threadIdx.x;
    float s = 0.f;
#pragma unroll
    for (int z = 0; z < NCHUNK; ++z) s += part[(size_t)z * (BB * HID) + t];
    meanf[t] = s * (1.0f / (float)SS);
}

__global__ __launch_bounds__(32)
void router_kernel(const float* __restrict__ meanf,
                   const float* __restrict__ Wr,
                   const float* __restrict__ br,
                   int* __restrict__ cnt) {
    int h = threadIdx.x;
    if (h >= HH) return;
    float acc = 0.f;
    for (int b = 0; b < BB; ++b) {
        float z = br[h];
        const float* m = meanf + (size_t)b * HID;
        for (int i = 0; i < HID; ++i) z += m[i] * Wr[(size_t)i * HH + h];
        acc += 1.0f / (1.0f + __expf(-z));
    }
    float head_score = acc * (1.0f / (float)BB);
    float eff = 0.5f * (1.0f - 0.5f * head_score);      // SR * (1 - hs/2)
    int c = (int)floorf((float)BSZ * eff);
    cnt[h] = c < 1 ? 1 : c;
}

// ---------------------------------------------------------------------------
// One-shot conversions: hidden f32 -> bf16;  W f32 [K][N] -> bf16 W^T [N][K]
// ---------------------------------------------------------------------------
__global__ __launch_bounds__(256)
void cvt_x_kernel(const float* __restrict__ x, unsigned short* __restrict__ xbf) {
    size_t base = ((size_t)blockIdx.x * 256 + threadIdx.x) * 16;
    su8 a, b;
#pragma unroll
    for (int j = 0; j < 8; ++j) { a[j] = f2bf(x[base + j]); b[j] = f2bf(x[base + 8 + j]); }
    *(su8*)&xbf[base]     = a;
    *(su8*)&xbf[base + 8] = b;
}

__global__ __launch_bounds__(256)
void cvt_wT_kernel(const float* __restrict__ W, unsigned short* __restrict__ WT) {
    __shared__ unsigned short Tl[64][72];
    const int k0 = blockIdx.x * 64;    // K tile
    const int n0 = blockIdx.y * 64;    // N tile
    const int tid = threadIdx.x;
    {
        int r  = tid >> 2;
        int cg = (tid & 3) * 16;
        const float* g = W + (size_t)(k0 + r) * HID + n0 + cg;
#pragma unroll
        for (int j = 0; j < 16; ++j) Tl[r][cg + j] = f2bf(g[j]);
    }
    __syncthreads();
    {
        int n  = tid >> 2;
        int kg = (tid & 3) * 16;
        su8 a, b;
#pragma unroll
        for (int j = 0; j < 8; ++j) { a[j] = Tl[kg + j][n]; b[j] = Tl[kg + 8 + j][n]; }
        unsigned short* o = WT + (size_t)(n0 + n) * HID + k0 + kg;
        *(su8*)o       = a;
        *(su8*)(o + 8) = b;
    }
}

// ---------------------------------------------------------------------------
// bf16 WMMA GEMM: C[M,N] = A[M,K] * B[K,N], with B given pre-transposed as
// Bt[N][K] bf16.  WG = 256 threads (8 waves); 128x128 tile; K-step 32;
// double-buffered LDS; prefetch 2 K-steps ahead.
// ---------------------------------------------------------------------------
template <bool OUT_BF16>
__global__ __launch_bounds__(256)
void gemm_bf(const unsigned short* __restrict__ A,
             const unsigned short* __restrict__ Bt,
             void* __restrict__ Cptr, int M, int N, int K) {
    __shared__ unsigned short Al[2][128][40];   // [buf][row][k]
    __shared__ unsigned short Bl[2][128][40];   // [buf][col][k]

    const int tid  = threadIdx.x;
    const int lane = tid & 31;
    const int wv   = tid >> 5;
    const int wr   = wv & 3;        // wave row  (32 rows)
    const int wc   = wv >> 2;       // wave col  (64 cols)
    const int m0   = blockIdx.y * 128;
    const int n0   = blockIdx.x * 128;
    const int r16  = lane & 15;
    const int hi   = lane >> 4;

    // staging addresses for this thread
    const int srow = tid >> 1;
    const int scg  = (tid & 1) * 16;
    const unsigned short* gA = A  + (size_t)(m0 + srow) * K + scg;
    const unsigned short* gB = Bt + (size_t)(n0 + srow) * K + scg;

    v8f acc[2][4] = {};

    const int nk = K / 32;
    // prologue: stage k-step 0 into buffer 0
    {
        *(su8*)&Al[0][srow][scg]     = *(const su8*)gA;
        *(su8*)&Al[0][srow][scg + 8] = *(const su8*)(gA + 8);
        *(su8*)&Bl[0][srow][scg]     = *(const su8*)gB;
        *(su8*)&Bl[0][srow][scg + 8] = *(const su8*)(gB + 8);
    }

    for (int it = 0; it < nk; ++it) {
        __syncthreads();
        // stage next k-step into the other buffer (overlaps with compute)
        if (it + 1 < nk) {
            const int nb = (it + 1) & 1;
            const unsigned short* pa = gA + (size_t)(it + 1) * 32;
            const unsigned short* pb = gB + (size_t)(it + 1) * 32;
            *(su8*)&Al[nb][srow][scg]     = *(const su8*)pa;
            *(su8*)&Al[nb][srow][scg + 8] = *(const su8*)(pa + 8);
            *(su8*)&Bl[nb][srow][scg]     = *(const su8*)pb;
            *(su8*)&Bl[nb][srow][scg + 8] = *(const su8*)(pb + 8);
            // speculative prefetch two k-steps ahead -> global_prefetch_b8
            __builtin_prefetch(gA + (size_t)(it + 2) * 32, 0, 0);
            __builtin_prefetch(gB + (size_t)(it + 2) * 32, 0, 0);
        }
        // compute on current buffer
        const int cb = it & 1;
        v16bf a[2], b[4];
#pragma unroll
        for (int rt = 0; rt < 2; ++rt) {
            int row = wr * 32 + rt * 16 + r16;
            int kb  = hi * 8;
            a[rt] = ldfrag(&Al[cb][row][kb], &Al[cb][row][16 + kb]);
        }
#pragma unroll
        for (int ct = 0; ct < 4; ++ct) {
            int col = wc * 64 + ct * 16 + r16;
            int kb  = hi * 8;
            b[ct] = ldfrag(&Bl[cb][col][kb], &Bl[cb][col][kb + 16]);
        }
#pragma unroll
        for (int rt = 0; rt < 2; ++rt)
#pragma unroll
            for (int ct = 0; ct < 4; ++ct)
                acc[rt][ct] = wmma_bf16(a[rt], b[ct], acc[rt][ct]);
    }

    // epilogue
#pragma unroll
    for (int rt = 0; rt < 2; ++rt) {
#pragma unroll
        for (int ct = 0; ct < 4; ++ct) {
            int col  = n0 + wc * 64 + ct * 16 + r16;
            int rowb = m0 + wr * 32 + rt * 16 + hi * 8;
#pragma unroll
            for (int r = 0; r < 8; ++r) {
                if (OUT_BF16)
                    ((unsigned short*)Cptr)[(size_t)(rowb + r) * N + col] =
                        f2bf(acc[rt][ct][r]);
                else
                    ((float*)Cptr)[(size_t)(rowb + r) * N + col] = acc[rt][ct][r];
            }
        }
    }
}

// ---------------------------------------------------------------------------
// Block attention: one WG per (nb, h, b). 64x64 block, D=128.
//   S = Q K^T * scale  (WMMA, frags straight from global bf16 q/k)
//   per-row variable top-k (stable-argsort rank < cnt[h]) -> mask to 0
//   softmax (4 threads per row), O = P V (WMMA, V transposed in LDS)
// ---------------------------------------------------------------------------
__global__ __launch_bounds__(256)
void attn_block_kernel(const unsigned short* __restrict__ q,
                       const unsigned short* __restrict__ k,
                       const unsigned short* __restrict__ v,
                       const int* __restrict__ cnt,
                       unsigned short* __restrict__ attn) {
    __shared__ float          Sf[64][68];    // raw scores
    __shared__ float          Mf[64][68];    // masked scores / exp scratch
    __shared__ unsigned short Pl[64][72];    // probabilities, bf16
    __shared__ unsigned short Vt[128][72];   // V transposed [d][row], bf16
    __shared__ float          redm[64][4];   // per-quarter max
    __shared__ float          reds[64][4];   // per-quarter sum

    const int nb = blockIdx.x, h = blockIdx.y, b = blockIdx.z;
    const int tid  = threadIdx.x;
    const int lane = tid & 31;
    const int wv   = tid >> 5;
    const int r16  = lane & 15;
    const int hi   = lane >> 4;
    const size_t rowbase = (size_t)b * SS + (size_t)nb * BSZ;
    const int    colb    = h * DD;

    // ---- stage V transposed: thread -> row tid/4, 32 d's
    {
        int r  = tid >> 2;
        int dg = (tid & 3) * 32;
        const unsigned short* vp = v + (rowbase + r) * HID + colb + dg;
#pragma unroll
        for (int c = 0; c < 4; ++c) {
            su8 x = *(const su8*)(vp + c * 8);
#pragma unroll
            for (int j = 0; j < 8; ++j) Vt[dg + c * 8 + j][r] = x[j];
        }
    }

    // ---- S = Q K^T : wave wv -> row-tile wv&3, col-tiles 2*(wv>>2)+{0,1}
    {
        const int rt  = wv & 3;
        const int ct0 = (wv >> 2) * 2;
        v8f sacc[2] = {};
#pragma unroll
        for (int kt = 0; kt < 4; ++kt) {
            int k0 = kt * 32;
            const unsigned short* qa =
                q + (rowbase + rt * 16 + r16) * HID + colb + k0 + hi * 8;
            v16bf af = ldfrag(qa, qa + 16);
#pragma unroll
            for (int ci = 0; ci < 2; ++ci) {
                // need K^T: row-major K is exactly the B-fragment layout
                const unsigned short* kb =
                    k + (rowbase + (ct0 + ci) * 16 + r16) * HID + colb + k0 + hi * 8;
                v16bf bf = ldfrag(kb, kb + 16);
                sacc[ci] = wmma_bf16(af, bf, sacc[ci]);
            }
        }
#pragma unroll
        for (int ci = 0; ci < 2; ++ci) {
            int col  = (ct0 + ci) * 16 + r16;
            int rowb = rt * 16 + hi * 8;
#pragma unroll
            for (int r = 0; r < 8; ++r) Sf[rowb + r][col] = sacc[ci][r] * SCALE;
        }
    }
    __syncthreads();

    // ---- top-k mask + softmax, 4 threads per row (each owns 16 columns)
    {
        const int row = tid & 63;
        const int qd  = tid >> 6;
        const int c   = cnt[h];
        const int j0  = qd * 16;
        float mx = 0.0f;                       // zeros always present in masked row
#pragma unroll
        for (int jj = 0; jj < 16; ++jj) {
            int   j  = j0 + jj;
            float sj = Sf[row][j];
            int rank = 0;
            for (int kk = 0; kk < BSZ; ++kk) {
                float sk = Sf[row][kk];
                rank += (sk > sj) || ((sk == sj) && (kk < j));
            }
            float m = (rank < c) ? sj : 0.0f;
            Mf[row][j] = m;
            mx = fmaxf(mx, m);
        }
        redm[row][qd] = mx;
        __syncthreads();
        float rmx = fmaxf(fmaxf(redm[row][0], redm[row][1]),
                          fmaxf(redm[row][2], redm[row][3]));
        float sum = 0.f;
#pragma unroll
        for (int jj = 0; jj < 16; ++jj) {
            int   j = j0 + jj;
            float e = __expf(Mf[row][j] - rmx);
            Mf[row][j] = e;
            sum += e;
        }
        reds[row][qd] = sum;
        __syncthreads();
        float rs   = reds[row][0] + reds[row][1] + reds[row][2] + reds[row][3];
        float rinv = 1.0f / rs;
#pragma unroll
        for (int jj = 0; jj < 16; ++jj) {
            int j = j0 + jj;
            Pl[row][j] = f2bf(Mf[row][j] * rinv);
        }
    }
    __syncthreads();

    // ---- O = P V : wave wv -> row-tile wv&3, col-tiles 4*(wv>>2)+{0..3}
    {
        const int rt = wv & 3;
        const int cq = wv >> 2;
        v8f oacc[4] = {};
#pragma unroll
        for (int kt = 0; kt < 2; ++kt) {
            int k0 = kt * 32;
            int row = rt * 16 + r16;
            int kb  = hi * 8;
            v16bf af = ldfrag(&Pl[row][k0 + kb], &Pl[row][k0 + kb + 16]);
#pragma unroll
            for (int ci = 0; ci < 4; ++ci) {
                int col = cq * 64 + ci * 16 + r16;
                int kb2 = k0 + hi * 8;
                v16bf bf = ldfrag(&Vt[col][kb2], &Vt[col][kb2 + 16]);
                oacc[ci] = wmma_bf16(af, bf, oacc[ci]);
            }
        }
#pragma unroll
        for (int ci = 0; ci < 4; ++ci) {
            int dcol = colb + cq * 64 + ci * 16 + r16;
            size_t rowg = rowbase + rt * 16 + hi * 8;
#pragma unroll
            for (int r = 0; r < 8; ++r)
                attn[(rowg + r) * HID + dcol] = f2bf(oacc[ci][r]);
        }
    }
}

// ---------------------------------------------------------------------------
// Launch
// ---------------------------------------------------------------------------
extern "C" void kernel_launch(void* const* d_in, const int* in_sizes, int n_in,
                              void* d_out, int out_size, void* d_ws, size_t ws_size,
                              hipStream_t stream) {
    const float* hs = (const float*)d_in[0];
    const float* Wq = (const float*)d_in[1];
    const float* Wk = (const float*)d_in[2];
    const float* Wv = (const float*)d_in[3];
    const float* Wo = (const float*)d_in[4];
    const float* Wr = (const float*)d_in[5];
    const float* br = (const float*)d_in[6];
    (void)in_sizes; (void)n_in; (void)out_size; (void)ws_size;

    const size_t mat = (size_t)MROWS * HID;   // 16M elements (activations)
    const size_t wmt = (size_t)HID * HID;     // 4M elements (weights)
    char* ws = (char*)d_ws;
    unsigned short* qbf = (unsigned short*)ws;
    unsigned short* kbf = qbf + mat;
    unsigned short* vbf = kbf + mat;
    unsigned short* abf = vbf + mat;
    unsigned short* xbf = abf + mat;
    unsigned short* WqT = xbf + mat;
    unsigned short* WkT = WqT + wmt;
    unsigned short* WvT = WkT + wmt;
    unsigned short* WoT = WvT + wmt;
    float* part  = (float*)(WoT + wmt);
    float* meanf = part + (size_t)NCHUNK * BB * HID;
    int*   cnt   = (int*)(meanf + BB * HID);

    // routing (tiny)
    colmean_part_kernel<<<dim3((BB * HID) / 256, NCHUNK), 256, 0, stream>>>(hs, part);
    colmean_reduce_kernel<<<(BB * HID) / 256, 256, 0, stream>>>(part, meanf);
    router_kernel<<<1, 32, 0, stream>>>(meanf, Wr, br, cnt);

    // one-shot conversions
    cvt_x_kernel<<<(int)(mat / (256 * 16)), 256, 0, stream>>>(hs, xbf);
    dim3 wt(HID / 64, HID / 64);
    cvt_wT_kernel<<<wt, 256, 0, stream>>>(Wq, WqT);
    cvt_wT_kernel<<<wt, 256, 0, stream>>>(Wk, WkT);
    cvt_wT_kernel<<<wt, 256, 0, stream>>>(Wv, WvT);
    cvt_wT_kernel<<<wt, 256, 0, stream>>>(Wo, WoT);

    // Q/K/V projections (all-bf16 GEMM, bf16 out)
    dim3 gg(HID / 128, MROWS / 128);
    gemm_bf<true><<<gg, 256, 0, stream>>>(xbf, WqT, qbf, MROWS, HID, HID);
    gemm_bf<true><<<gg, 256, 0, stream>>>(xbf, WkT, kbf, MROWS, HID, HID);
    gemm_bf<true><<<gg, 256, 0, stream>>>(xbf, WvT, vbf, MROWS, HID, HID);

    // block-diagonal attention
    attn_block_kernel<<<dim3(NBL, HH, BB), 256, 0, stream>>>(qbf, kbf, vbf, cnt, abf);

    // output projection (bf16 in, f32 out)
    gemm_bf<false><<<gg, 256, 0, stream>>>(abf, WoT, (float*)d_out, MROWS, HID, HID);
}
